// DeltaGraph_79688823210239
// MI455X (gfx1250) — compile-verified, matrix operand order checked by reading, not run on hardware
//
#include <hip/hip_runtime.h>
#include <hip/hip_bf16.h>
#include <stdint.h>

// ---------------- problem constants ----------------
#define B_SZ   2
#define N_PTS  32768
#define M_PTS  2622      // ceil(0.08 * 32768)
#define M_PAD  2688      // multiple of 128 for WMMA tiles
#define HID    256
#define HEADS  4
#define HDIM   64
#define KNB    16
#define IN_K   35
#define IN_KP  64        // padded K for input projection

typedef __attribute__((ext_vector_type(16))) _Float16 v16h;
typedef __attribute__((ext_vector_type(8)))  _Float16 v8h;
typedef __attribute__((ext_vector_type(8)))  float    v8f;

// =====================================================================
// WMMA GEMM:  C[M x N] = A[M x K](f16 row-major) * B(f16, stored as
// Bt[N x K] i.e. K-major / column-major of B) with f32 accumulation.
// Epilogue: +bias, +residual, optional ReLU; writes f32 and/or f16.
// Block = 256 threads = 8 waves arranged 4x2 -> 128x64 tile,
// each wave owns a 32x32 tile = 4 v_wmma_f32_16x16x32_f16 per k-step.
// M must be a multiple of 128, N of 64, K of 32 (buffers are padded).
// =====================================================================
__device__ __forceinline__ v16h dg_load_frag(const _Float16* __restrict__ base,
                                             int ld, int row0, int k0, int lane)
{
  const int r = lane & 15;
  const int h = lane >> 4;
  const _Float16* p = base + (size_t)(row0 + r) * ld + k0 + h * 8;
  v8h lo = *(const v8h*)(p);        // K = k0 + h*8 .. +7   -> elements 0..7
  v8h hi = *(const v8h*)(p + 16);   // K = k0 + 16 + h*8 .. -> elements 8..15
  v16h f;
#pragma unroll
  for (int i = 0; i < 8; ++i) { f[i] = lo[i]; f[i + 8] = hi[i]; }
  return f;
}

__global__ __launch_bounds__(256)
void dg_gemm(const _Float16* __restrict__ A, const _Float16* __restrict__ Bt,
             const float* __restrict__ bias, const float* __restrict__ resid,
             float* __restrict__ outF, _Float16* __restrict__ outH,
             int K, int N, int relu)
{
  const int lane = threadIdx.x & 31;
  const int wave = threadIdx.x >> 5;     // 0..7
  const int wm   = wave & 3;             // 4 waves along M
  const int wn   = wave >> 2;            // 2 waves along N
  const int rowBase = blockIdx.y * 128 + wm * 32;
  const int colBase = blockIdx.x * 64  + wn * 32;

  v8f zero = {0.f,0.f,0.f,0.f,0.f,0.f,0.f,0.f};
  v8f acc[2][2];
#pragma unroll
  for (int i = 0; i < 2; ++i)
#pragma unroll
    for (int j = 0; j < 2; ++j) acc[i][j] = zero;

  for (int k0 = 0; k0 < K; k0 += 32) {
    if (k0 + 32 < K) {  // hint next A tile -> global_prefetch_b8
      __builtin_prefetch((const void*)(A + (size_t)(rowBase + (lane & 15)) * K + k0 + 32), 0, 0);
    }
    v16h a0 = dg_load_frag(A,  K, rowBase,      k0, lane);
    v16h a1 = dg_load_frag(A,  K, rowBase + 16, k0, lane);
    v16h b0 = dg_load_frag(Bt, K, colBase,      k0, lane);
    v16h b1 = dg_load_frag(Bt, K, colBase + 16, k0, lane);
    acc[0][0] = __builtin_amdgcn_wmma_f32_16x16x32_f16(false, a0, false, b0, (short)0, acc[0][0], false, false);
    acc[0][1] = __builtin_amdgcn_wmma_f32_16x16x32_f16(false, a0, false, b1, (short)0, acc[0][1], false, false);
    acc[1][0] = __builtin_amdgcn_wmma_f32_16x16x32_f16(false, a1, false, b0, (short)0, acc[1][0], false, false);
    acc[1][1] = __builtin_amdgcn_wmma_f32_16x16x32_f16(false, a1, false, b1, (short)0, acc[1][1], false, false);
  }

  // C layout: VGPR t -> M = t + 8*(lane>>4), N = lane&15
  const int cn   = lane & 15;
  const int half = lane >> 4;
#pragma unroll
  for (int i = 0; i < 2; ++i)
#pragma unroll
    for (int j = 0; j < 2; ++j) {
      const int col = colBase + j * 16 + cn;
#pragma unroll
      for (int t = 0; t < 8; ++t) {
        const int row = rowBase + i * 16 + half * 8 + t;
        const size_t off = (size_t)row * N + col;
        float c = acc[i][j][t];
        if (bias)  c += bias[col];
        if (resid) c += resid[off];
        if (relu)  c = fmaxf(c, 0.f);
        if (outF) outF[off] = c;
        if (outH) outH[off] = (_Float16)c;
      }
    }
}

// =====================================================================
// Weight convert: fp32 row-major W[K x N]  ->  f16 Bt[N x Kpad] (zero pad)
// =====================================================================
__global__ void dg_convw(const float* __restrict__ W, _Float16* __restrict__ Wc,
                         int K, int N, int Kpad)
{
  int t = blockIdx.x * blockDim.x + threadIdx.x;
  if (t >= N * Kpad) return;
  int n = t / Kpad, k = t % Kpad;
  Wc[(size_t)n * Kpad + k] = (k < K) ? (_Float16)W[(size_t)k * N + n] : (_Float16)0.f;
}

// =====================================================================
// Build padded input features  [pos(3) | vel_flat(15) | t(10) | geom(7) | 0pad]
// =====================================================================
__global__ void dg_build_x0(const float* __restrict__ pos, const float* __restrict__ velin,
                            const float* __restrict__ tvec, const float* __restrict__ geom,
                            _Float16* __restrict__ A0, int N, int total)
{
  int gt = blockIdx.x * blockDim.x + threadIdx.x;
  if (gt >= total) return;
  int b = gt / N, n = gt % N;
  _Float16* a = A0 + (size_t)gt * IN_KP;
#pragma unroll
  for (int c = 0; c < 3; ++c) a[c] = (_Float16)pos[(size_t)gt * 3 + c];
#pragma unroll
  for (int ti = 0; ti < 5; ++ti)
#pragma unroll
    for (int c = 0; c < 3; ++c)
      a[3 + ti * 3 + c] = (_Float16)velin[(((size_t)b * 5 + ti) * N + n) * 3 + c];
#pragma unroll
  for (int k = 0; k < 10; ++k) a[18 + k] = (_Float16)tvec[b * 10 + k];
#pragma unroll
  for (int k = 0; k < 7;  ++k) a[28 + k] = (_Float16)geom[(size_t)gt * 7 + k];
#pragma unroll
  for (int k = IN_K; k < IN_KP; ++k) a[k] = (_Float16)0.f;
}

// =====================================================================
// LayerNorm over 256 cols, one wave32 per row (8 rows per 256-thread block)
// =====================================================================
__global__ void dg_ln(const float* __restrict__ X, const float* __restrict__ g,
                      const float* __restrict__ bb, float* __restrict__ outF,
                      _Float16* __restrict__ outH, int rows, int relu)
{
  int lane = threadIdx.x & 31;
  int row  = blockIdx.x * 8 + (threadIdx.x >> 5);
  if (row >= rows) return;
  const float* x = X + (size_t)row * HID;
  float v[8];
  float s = 0.f;
#pragma unroll
  for (int k = 0; k < 8; ++k) { v[k] = x[lane + 32 * k]; s += v[k]; }
#pragma unroll
  for (int o = 16; o > 0; o >>= 1) s += __shfl_xor(s, o, 32);
  float mu = s * (1.f / 256.f);
  float q = 0.f;
#pragma unroll
  for (int k = 0; k < 8; ++k) { float d = v[k] - mu; q += d * d; }
#pragma unroll
  for (int o = 16; o > 0; o >>= 1) q += __shfl_xor(q, o, 32);
  float inv = rsqrtf(q * (1.f / 256.f) + 1e-5f);
#pragma unroll
  for (int k = 0; k < 8; ++k) {
    int c = lane + 32 * k;
    float o = (v[k] - mu) * inv * g[c] + bb[c];
    if (relu) o = fmaxf(o, 0.f);
    size_t off = (size_t)row * HID + c;
    if (outF) outF[off] = o;
    if (outH) outH[off] = (_Float16)o;
  }
}

// =====================================================================
// Farthest point sampling: one block per sample, sequential M steps.
// =====================================================================
#define FPS_T 256
__global__ void dg_fps(const float* __restrict__ pos, float* __restrict__ mind,
                       int* __restrict__ outIdx, int N, int M)
{
  int b = blockIdx.x;
  const float* P = pos + (size_t)b * N * 3;
  float* md = mind + (size_t)b * N;
  __shared__ float sv[FPS_T];
  __shared__ int   si[FPS_T];
  __shared__ int   s_cur;
  int tid = threadIdx.x;
  for (int i = tid; i < N; i += FPS_T) md[i] = 1e30f;
  if (tid == 0) s_cur = 0;
  __syncthreads();
  for (int m = 0; m < M; ++m) {
    int cur = s_cur;
    if (tid == 0) outIdx[(size_t)b * M + m] = cur;
    float cx = P[cur * 3], cy = P[cur * 3 + 1], cz = P[cur * 3 + 2];
    float bv = -1.f; int bi = 0;
    for (int i = tid; i < N; i += FPS_T) {
      float dx = P[i * 3] - cx, dy = P[i * 3 + 1] - cy, dz = P[i * 3 + 2] - cz;
      float d = dx * dx + dy * dy + dz * dz;
      float nm = fminf(md[i], d);
      md[i] = nm;
      if (nm > bv) { bv = nm; bi = i; }
    }
    sv[tid] = bv; si[tid] = bi;
    __syncthreads();
    for (int s = FPS_T / 2; s > 0; s >>= 1) {
      if (tid < s && sv[tid + s] > sv[tid]) { sv[tid] = sv[tid + s]; si[tid] = si[tid + s]; }
      __syncthreads();
    }
    if (tid == 0) s_cur = si[0];
    __syncthreads();
  }
}

// =====================================================================
// Gather coarse level: xcur (f32, zero pad rows) and pos_l1
// one block per padded row, 256 threads = channels
// =====================================================================
__global__ void dg_gather(const float* __restrict__ X, const float* __restrict__ pos,
                          const int* __restrict__ idx, float* __restrict__ xcur,
                          float* __restrict__ posL, int b, int N, int M)
{
  int i = blockIdx.x, c = threadIdx.x;
  float v = 0.f;
  if (i < M) {
    int s = idx[i];
    v = X[((size_t)b * N + s) * HID + c];
    if (c < 3) posL[i * 3 + c] = pos[((size_t)b * N + s) * 3 + c];
  }
  xcur[(size_t)i * HID + c] = v;
}

// =====================================================================
// 16-NN on coarse points (pos_l1 staged in LDS), one thread per row
// =====================================================================
__global__ void dg_knn16(const float* __restrict__ posL, int* __restrict__ nbr, int M)
{
  __shared__ float sp[M_PTS * 3];
  for (int i = threadIdx.x; i < M * 3; i += blockDim.x) sp[i] = posL[i];
  __syncthreads();
  int i = blockIdx.x * blockDim.x + threadIdx.x;
  if (i >= M) return;
  float px = sp[i * 3], py = sp[i * 3 + 1], pz = sp[i * 3 + 2];
  float nd[KNB]; int ni[KNB];
#pragma unroll
  for (int k = 0; k < KNB; ++k) { nd[k] = 1e30f; ni[k] = 0; }
  for (int j = 0; j < M; ++j) {
    if (j == i) continue;
    float dx = sp[j * 3] - px, dy = sp[j * 3 + 1] - py, dz = sp[j * 3 + 2] - pz;
    float d = dx * dx + dy * dy + dz * dz;
    if (d < nd[KNB - 1]) {
      int k = KNB - 1;
      while (k > 0 && nd[k - 1] > d) { nd[k] = nd[k - 1]; ni[k] = ni[k - 1]; --k; }
      nd[k] = d; ni[k] = j;
    }
  }
#pragma unroll
  for (int k = 0; k < KNB; ++k) nbr[(size_t)i * KNB + k] = ni[k];
}

// =====================================================================
// Edge attention weights (softmax over 16 contiguous edges per node/head)
// =====================================================================
__global__ void dg_attn(const float* __restrict__ Q, const float* __restrict__ Km,
                        const float* __restrict__ posL, const int* __restrict__ nbr,
                        const float* __restrict__ we, const float* __restrict__ be,
                        float* __restrict__ aw, int M)
{
  int t = blockIdx.x * blockDim.x + threadIdx.x;
  int i = t >> 2, h = t & 3;
  if (i >= M) return;
  float px = posL[i * 3], py = posL[i * 3 + 1], pz = posL[i * 3 + 2];
  const float* q = Q + (size_t)i * HID + h * HDIM;
  float logit[KNB];
  float mx = -1e30f;
#pragma unroll 1
  for (int j = 0; j < KNB; ++j) {
    int s = nbr[(size_t)i * KNB + j];
    float rx = posL[s * 3] - px, ry = posL[s * 3 + 1] - py, rz = posL[s * 3 + 2] - pz;
    float dist = sqrtf(rx * rx + ry * ry + rz * rz);
    float eb = rx * we[0 * HEADS + h] + ry * we[1 * HEADS + h] +
               rz * we[2 * HEADS + h] + dist * we[3 * HEADS + h] + be[h];
    const float* kk = Km + (size_t)s * HID + h * HDIM;
    float dot = 0.f;
#pragma unroll
    for (int d = 0; d < HDIM; ++d) dot += q[d] * kk[d];
    float a = dot * 0.125f + eb;   // HEAD_DIM^-0.5 = 1/8
    logit[j] = a;
    mx = fmaxf(mx, a);
  }
  float sum = 0.f;
#pragma unroll
  for (int j = 0; j < KNB; ++j) { logit[j] = __expf(logit[j] - mx); sum += logit[j]; }
  sum = fmaxf(sum, 1e-6f);
  float rs = 1.f / sum;
#pragma unroll
  for (int j = 0; j < KNB; ++j) aw[((size_t)i * KNB + j) * HEADS + h] = logit[j] * rs;
}

// =====================================================================
// Aggregate weighted V -> f16 (A operand of the O-projection GEMM)
// one block per padded row, 256 threads = channels
// =====================================================================
__global__ void dg_agg(const float* __restrict__ V, const float* __restrict__ aw,
                       const int* __restrict__ nbr, _Float16* __restrict__ aggH, int M)
{
  int i = blockIdx.x, c = threadIdx.x;
  float acc = 0.f;
  if (i < M) {
    int h = c >> 6;
#pragma unroll 1
    for (int j = 0; j < KNB; ++j) {
      int s = nbr[(size_t)i * KNB + j];
      acc += aw[((size_t)i * KNB + j) * HEADS + h] * V[(size_t)s * HID + c];
    }
  }
  aggH[(size_t)i * HID + c] = (_Float16)acc;
}

// =====================================================================
// 3-NN interpolation weights (coarse pos in LDS), one thread per fine pt
// =====================================================================
__global__ void dg_knn3(const float* __restrict__ posF, const float* __restrict__ posL,
                        int* __restrict__ idx3, float* __restrict__ w3, int N, int M)
{
  __shared__ float sp[M_PTS * 3];
  for (int i = threadIdx.x; i < M * 3; i += blockDim.x) sp[i] = posL[i];
  __syncthreads();
  int n = blockIdx.x * blockDim.x + threadIdx.x;
  if (n >= N) return;
  float px = posF[(size_t)n * 3], py = posF[(size_t)n * 3 + 1], pz = posF[(size_t)n * 3 + 2];
  float d0 = 1e30f, d1 = 1e30f, d2 = 1e30f;
  int i0 = 0, i1 = 0, i2 = 0;
  for (int j = 0; j < M; ++j) {
    float dx = sp[j * 3] - px, dy = sp[j * 3 + 1] - py, dz = sp[j * 3 + 2] - pz;
    float d = dx * dx + dy * dy + dz * dz;
    if (d < d0)      { d2 = d1; i2 = i1; d1 = d0; i1 = i0; d0 = d; i0 = j; }
    else if (d < d1) { d2 = d1; i2 = i1; d1 = d;  i1 = j; }
    else if (d < d2) { d2 = d;  i2 = j; }
  }
  float w0 = 1.f / fmaxf(sqrtf(fmaxf(d0, 0.f)), 1e-8f);
  float w1 = 1.f / fmaxf(sqrtf(fmaxf(d1, 0.f)), 1e-8f);
  float w2 = 1.f / fmaxf(sqrtf(fmaxf(d2, 0.f)), 1e-8f);
  float rs = 1.f / fmaxf(w0 + w1 + w2, 1e-8f);
  idx3[(size_t)n * 3] = i0; idx3[(size_t)n * 3 + 1] = i1; idx3[(size_t)n * 3 + 2] = i2;
  w3[(size_t)n * 3] = w0 * rs; w3[(size_t)n * 3 + 1] = w1 * rs; w3[(size_t)n * 3 + 2] = w2 * rs;
}

// =====================================================================
// concat [x_s | x_up] as f16 rows of 512 (A operand of fuse GEMM)
// =====================================================================
__global__ void dg_concat(const _Float16* __restrict__ Xh, const float* __restrict__ xl1,
                          const int* __restrict__ idx3, const float* __restrict__ w3,
                          _Float16* __restrict__ catH, int b, int N)
{
  int t = blockIdx.x * blockDim.x + threadIdx.x;
  if (t >= N * HID) return;
  int n = t / HID, c = t % HID;
  size_t p = (size_t)b * N + n;
  catH[p * 512 + c] = Xh[p * HID + c];
  float up = 0.f;
#pragma unroll
  for (int k = 0; k < 3; ++k)
    up += w3[(size_t)n * 3 + k] * xl1[(size_t)idx3[(size_t)n * 3 + k] * HID + c];
  catH[p * 512 + HID + c] = (_Float16)up;
}

// =====================================================================
// out projection 256 -> 15 (too narrow for a 64-wide WMMA tile; scalar)
// =====================================================================
__global__ void dg_outproj(const float* __restrict__ Z, const float* __restrict__ W,
                           const float* __restrict__ bias, float* __restrict__ delta, int P)
{
  int t = blockIdx.x * blockDim.x + threadIdx.x;
  if (t >= P * 15) return;
  int p = t / 15, o = t % 15;
  float acc = bias[o];
  const float* z = Z + (size_t)p * HID;
#pragma unroll 8
  for (int c = 0; c < HID; ++c) acc += z[c] * W[c * 15 + o];
  delta[(size_t)p * 15 + o] = acc;
}

__global__ void dg_mask_init(int* __restrict__ mask, int n)
{
  int t = blockIdx.x * blockDim.x + threadIdx.x;
  if (t < n) mask[t] = 0;
}
__global__ void dg_mask_set(const int* __restrict__ idcs, int* __restrict__ mask, int N, int total)
{
  int t = blockIdx.x * blockDim.x + threadIdx.x;
  if (t >= total) return;
  int b = t / 512;
  mask[(size_t)b * N + idcs[t]] = 1;
}

// =====================================================================
// baseline extrapolation + delta + airfoil mask
// =====================================================================
__global__ void dg_final(const float* __restrict__ tvec, const float* __restrict__ velin,
                         const float* __restrict__ delta, const int* __restrict__ mask,
                         float* __restrict__ out, int N, int total)
{
  int gt = blockIdx.x * blockDim.x + threadIdx.x;
  if (gt >= total) return;
  int b = gt / N, n = gt % N;
  const float* tb = tvec + b * 10;
  float dt = fmaxf(tb[4] - tb[3], 1e-6f);
  float m = mask[gt] ? 0.f : 1.f;
#pragma unroll
  for (int c = 0; c < 3; ++c) {
    float vlast = velin[(((size_t)b * 5 + 4) * N + n) * 3 + c];
    float vprev = velin[(((size_t)b * 5 + 3) * N + n) * 3 + c];
    float slope = (vlast - vprev) / dt;
#pragma unroll
    for (int to = 0; to < 5; ++to) {
      float base = vlast + slope * (tb[5 + to] - tb[4]);
      float d = delta[((size_t)b * N + n) * 15 + to * 3 + c];
      out[(((size_t)b * 5 + to) * N + n) * 3 + c] = (base + d) * m;
    }
  }
}

// =====================================================================
// host orchestration
// =====================================================================
extern "C" void kernel_launch(void* const* d_in, const int* in_sizes, int n_in,
                              void* d_out, int out_size, void* d_ws, size_t ws_size,
                              hipStream_t stream)
{
  (void)in_sizes; (void)n_in; (void)out_size; (void)ws_size;
  const int B = B_SZ, N = N_PTS, M = M_PTS, P = B_SZ * N_PTS;

  const float* t_in  = (const float*)d_in[0];
  const float* pos   = (const float*)d_in[1];
  const int*   idcs  = (const int*)d_in[2];
  const float* velin = (const float*)d_in[3];
  const float* geom  = (const float*)d_in[4];

  // params: JAX pytree flatten order (sorted dict keys; blocks list in order)
  struct Blk { const float *be,*bo,*f1_b,*f1_w,*f2_b,*f2_w,*n1_b,*n1_g,*n2_b,*n2_g,*we,*wk,*wo,*wq,*wv; } blk[2];
  for (int i = 0; i < 2; ++i) {
    int o = 5 + i * 15;
    blk[i].be   = (const float*)d_in[o + 0];  blk[i].bo   = (const float*)d_in[o + 1];
    blk[i].f1_b = (const float*)d_in[o + 2];  blk[i].f1_w = (const float*)d_in[o + 3];
    blk[i].f2_b = (const float*)d_in[o + 4];  blk[i].f2_w = (const float*)d_in[o + 5];
    blk[i].n1_b = (const float*)d_in[o + 6];  blk[i].n1_g = (const float*)d_in[o + 7];
    blk[i].n2_b = (const float*)d_in[o + 8];  blk[i].n2_g = (const float*)d_in[o + 9];
    blk[i].we   = (const float*)d_in[o + 10]; blk[i].wk   = (const float*)d_in[o + 11];
    blk[i].wo   = (const float*)d_in[o + 12]; blk[i].wq   = (const float*)d_in[o + 13];
    blk[i].wv   = (const float*)d_in[o + 14];
  }
  const float* fuse_b = (const float*)d_in[35];
  const float* fuse_w = (const float*)d_in[36];
  const float* in_b   = (const float*)d_in[37];
  const float* in_w   = (const float*)d_in[38];
  const float* mid_b  = (const float*)d_in[39];
  const float* mid_w  = (const float*)d_in[40];
  const float* nfu_b  = (const float*)d_in[41];
  const float* nfu_g  = (const float*)d_in[42];
  const float* nin_b  = (const float*)d_in[43];
  const float* nin_g  = (const float*)d_in[44];
  const float* nmi_b  = (const float*)d_in[45];
  const float* nmi_g  = (const float*)d_in[46];
  const float* out_b  = (const float*)d_in[47];
  const float* out_w  = (const float*)d_in[48];

  // ---- workspace bump allocator ----
  char* wsp = (char*)d_ws;
  size_t off = 0;
  auto alloc = [&](size_t bytes) -> void* {
    void* p = wsp + off;
    off = (off + bytes + 255) & ~(size_t)255;
    return p;
  };

  _Float16* inC   = (_Float16*)alloc((size_t)HID * IN_KP * 2);
  _Float16 *wqC[2], *wkC[2], *wvC[2], *woC[2], *f1C[2], *f2C[2];
  for (int i = 0; i < 2; ++i) {
    wqC[i] = (_Float16*)alloc((size_t)HID * HID * 2);
    wkC[i] = (_Float16*)alloc((size_t)HID * HID * 2);
    wvC[i] = (_Float16*)alloc((size_t)HID * HID * 2);
    woC[i] = (_Float16*)alloc((size_t)HID * HID * 2);
    f1C[i] = (_Float16*)alloc((size_t)512 * HID * 2);   // Bt[512][256]
    f2C[i] = (_Float16*)alloc((size_t)HID * 512 * 2);   // Bt[256][512]
  }
  _Float16* fuseC = (_Float16*)alloc((size_t)HID * 512 * 2);
  _Float16* midC  = (_Float16*)alloc((size_t)HID * HID * 2);

  _Float16* A0   = (_Float16*)alloc((size_t)P * IN_KP * 2);
  float*    Xf   = (float*)   alloc((size_t)P * HID * 4);
  _Float16* Xh   = (_Float16*)alloc((size_t)P * HID * 2);
  _Float16* catH = (_Float16*)alloc((size_t)P * 512 * 2);
  float*    Yf   = (float*)   alloc((size_t)P * HID * 4);
  _Float16* Yh   = (_Float16*)alloc((size_t)P * HID * 2);
  float*    Zf   = (float*)   alloc((size_t)P * HID * 4);
  float*    delta= (float*)   alloc((size_t)P * 15 * 4);
  int*      mask = (int*)     alloc((size_t)P * 4);

  float*    mind = (float*)alloc((size_t)B * N * 4);
  int*      l1i  = (int*)  alloc((size_t)B * M * 4);
  float*    posL = (float*)alloc((size_t)M_PAD * 3 * 4);
  float*    xcur = (float*)alloc((size_t)M_PAD * HID * 4);
  _Float16* hH   = (_Float16*)alloc((size_t)M_PAD * HID * 2);
  float*    Qf   = (float*)alloc((size_t)M_PAD * HID * 4);
  float*    Kf   = (float*)alloc((size_t)M_PAD * HID * 4);
  float*    Vf   = (float*)alloc((size_t)M_PAD * HID * 4);
  _Float16* aggH = (_Float16*)alloc((size_t)M_PAD * HID * 2);
  _Float16* ffnH = (_Float16*)alloc((size_t)M_PAD * 512 * 2);
  float*    aw   = (float*)alloc((size_t)M_PAD * KNB * HEADS * 4);
  int*      nbr  = (int*)  alloc((size_t)M_PAD * KNB * 4);
  int*      idx3 = (int*)  alloc((size_t)N * 3 * 4);
  float*    w3   = (float*)alloc((size_t)N * 3 * 4);

  auto convw = [&](const float* W, _Float16* Wc, int K, int Nn, int Kp) {
    int tot = Nn * Kp;
    dg_convw<<<(tot + 255) / 256, 256, 0, stream>>>(W, Wc, K, Nn, Kp);
  };
  auto gemm = [&](const _Float16* A, const _Float16* Bt, const float* bias,
                  const float* resid, float* outF, _Float16* outH,
                  int Mr, int K, int Nn, int relu) {
    dim3 g(Nn / 64, Mr / 128);
    dg_gemm<<<g, 256, 0, stream>>>(A, Bt, bias, resid, outF, outH, K, Nn, relu);
  };

  // ---- weight conversion (fp32 -> f16, K-major) ----
  convw(in_w, inC, IN_K, HID, IN_KP);
  for (int i = 0; i < 2; ++i) {
    convw(blk[i].wq, wqC[i], HID, HID, HID);
    convw(blk[i].wk, wkC[i], HID, HID, HID);
    convw(blk[i].wv, wvC[i], HID, HID, HID);
    convw(blk[i].wo, woC[i], HID, HID, HID);
    convw(blk[i].f1_w, f1C[i], HID, 512, HID);
    convw(blk[i].f2_w, f2C[i], 512, HID, 512);
  }
  convw(fuse_w, fuseC, 512, HID, 512);
  convw(mid_w, midC, HID, HID, HID);

  // ---- input projection + LN + ReLU ----
  dg_build_x0<<<(P + 255) / 256, 256, 0, stream>>>(pos, velin, t_in, geom, A0, N, P);
  gemm(A0, inC, in_b, nullptr, Xf, nullptr, P, IN_KP, HID, 0);
  dg_ln<<<P / 8, 256, 0, stream>>>(Xf, nin_g, nin_b, Xf, Xh, P, 1);

  // ---- FPS (both samples in parallel, sequential inside) ----
  dg_fps<<<B, FPS_T, 0, stream>>>(pos, mind, l1i, N, M);

  // ---- per-sample coarse transformer + interpolation ----
  for (int b = 0; b < B; ++b) {
    dg_gather<<<M_PAD, 256, 0, stream>>>(Xf, pos, l1i + (size_t)b * M, xcur, posL, b, N, M);
    dg_knn16<<<(M + 255) / 256, 256, 0, stream>>>(posL, nbr, M);
    for (int L = 0; L < 2; ++L) {
      const Blk& pp = blk[L];
      dg_ln<<<M_PAD / 8, 256, 0, stream>>>(xcur, pp.n1_g, pp.n1_b, nullptr, hH, M_PAD, 0);
      gemm(hH, wqC[L], nullptr, nullptr, Qf, nullptr, M_PAD, HID, HID, 0);
      gemm(hH, wkC[L], nullptr, nullptr, Kf, nullptr, M_PAD, HID, HID, 0);
      gemm(hH, wvC[L], nullptr, nullptr, Vf, nullptr, M_PAD, HID, HID, 0);
      dg_attn<<<(M * HEADS + 255) / 256, 256, 0, stream>>>(Qf, Kf, posL, nbr, pp.we, pp.be, aw, M);
      dg_agg<<<M_PAD, 256, 0, stream>>>(Vf, aw, nbr, aggH, M);
      gemm(aggH, woC[L], pp.bo, xcur, xcur, nullptr, M_PAD, HID, HID, 0);
      dg_ln<<<M_PAD / 8, 256, 0, stream>>>(xcur, pp.n2_g, pp.n2_b, nullptr, hH, M_PAD, 0);
      gemm(hH, f1C[L], pp.f1_b, nullptr, nullptr, ffnH, M_PAD, HID, 512, 1);
      gemm(ffnH, f2C[L], pp.f2_b, xcur, xcur, nullptr, M_PAD, 512, HID, 0);
    }
    dg_knn3<<<(N + 255) / 256, 256, 0, stream>>>(pos + (size_t)b * N * 3, posL, idx3, w3, N, M);
    dg_concat<<<N, 256, 0, stream>>>(Xh, xcur, idx3, w3, catH, b, N);
  }

  // ---- fuse -> mid -> out ----
  gemm(catH, fuseC, fuse_b, nullptr, Yf, nullptr, P, 512, HID, 0);
  dg_ln<<<P / 8, 256, 0, stream>>>(Yf, nfu_g, nfu_b, nullptr, Yh, P, 1);
  gemm(Yh, midC, mid_b, nullptr, Zf, nullptr, P, HID, HID, 0);
  dg_ln<<<P / 8, 256, 0, stream>>>(Zf, nmi_g, nmi_b, Zf, nullptr, P, 1);
  dg_outproj<<<(P * 15 + 255) / 256, 256, 0, stream>>>(Zf, out_w, out_b, delta, P);

  // ---- mask + final combine ----
  dg_mask_init<<<(P + 255) / 256, 256, 0, stream>>>(mask, P);
  dg_mask_set<<<(B * 512 + 255) / 256, 256, 0, stream>>>(idcs, mask, N, B * 512);
  dg_final<<<(P + 255) / 256, 256, 0, stream>>>(t_in, velin, delta, mask, (float*)d_out, N, P);
}
